// Model_Side_lobe_gain_23785528886029
// MI455X (gfx1250) — matrix operand order repaired and checked by reading, not compiled, not run on hardware
//
#include <hip/hip_runtime.h>
#include <cstdint>
#include <cstddef>

// ---------------------------------------------------------------------------
// CDNA5 (gfx1250) bf16 WMMA implementation of the side-lobe loss.
//   Phase 1: w_full (bf16, mirrored/normalized, 128x1024) ; svpT = (theta*sv)^T (bf16, [2][128][16384])
//   Phase 2: B = G @ (theta*sv)  via v_wmma_f32_16x16x32_bf16, K-split x8, double-buffered LDS
//   Phase 3: deterministic reduce -> Bt bf16 ;  sout = |w_full @ B|^2/(A*L)
//   Phase 4: scalar stats (diag, band, unbiased std) -> d_out[0]
// Complex GEMM uses 2 accumulators per tile (re, im) by pre-negating the A-imag
// fragment (exact bf16 signflip), so each WMMA chain accumulates in place.
// ---------------------------------------------------------------------------

typedef __attribute__((ext_vector_type(16))) __bf16 bf16x16;
typedef __attribute__((ext_vector_type(8)))  float  f32x8;

#define A_ROWS 64
#define B_ROWS 128
#define LDS_STRIDE 56                       // halves/row: 112B, 16B-aligned, conflict-free
#define LDS_A_HALVES (2*A_ROWS*LDS_STRIDE)  // 7168
#define LDS_B_HALVES (2*B_ROWS*LDS_STRIDE)  // 14336
#define LDS_TOTAL    (LDS_A_HALVES + LDS_B_HALVES)       // 21504 halves = 43008 B
#define LDS_BYTES_2BUF (2 * LDS_TOTAL * 2)               // 86016 B dynamic LDS

// workspace layout (bytes)
#define OFF_SVPT   ((size_t)0)              // bf16 [2][128][16384]  = 8 MB
#define OFF_WFULL  ((size_t)8388608)        // bf16 [2][128][1024]   = 512 KB
#define OFF_PART   ((size_t)8912896)        // f32  [2][8][128][1024]= 8 MB
#define OFF_BT     ((size_t)17301504)       // bf16 [2][128][1024]   = 512 KB
#define OFF_SOUT   ((size_t)17825792)       // f32  [128][128]       = 64 KB

__device__ __forceinline__ unsigned f2bf(float f) {
  unsigned u = __float_as_uint(f);
  return ((u + 0x7FFFu + ((u >> 16) & 1u)) >> 16) & 0xFFFFu;  // RNE f32->bf16
}

union FragU { uint4 q[2]; bf16x16 v; };

// A-matrix 16x32 bf16 fragment: lane l holds row M=l&15; K = e + (l&16) (16 contiguous halves)
__device__ __forceinline__ bf16x16 ld_fragA(const unsigned short* lds, int mat, int strip, int lane) {
  int row = strip * 16 + (lane & 15);
  int klo = lane & 16;
  const unsigned short* p = lds + (mat * A_ROWS + row) * LDS_STRIDE;
  FragU f;
  f.q[0] = *(const uint4*)(p + klo);
  f.q[1] = *(const uint4*)(p + klo + 8);
  return f.v;
}

// B-matrix 32x16 bf16 fragment: lane l holds col N=l&15; K = e + (l&16) (16 contiguous halves)
__device__ __forceinline__ bf16x16 ld_fragB(const unsigned short* lds, int mat, int nrow, int lane) {
  int row = nrow + (lane & 15);
  int kb  = lane & 16;
  const unsigned short* p = lds + LDS_A_HALVES + (mat * B_ROWS + row) * LDS_STRIDE;
  FragU f;
  f.q[0] = *(const uint4*)(p + kb);
  f.q[1] = *(const uint4*)(p + kb + 8);
  return f.v;
}

__device__ __forceinline__ bf16x16 neg_frag(bf16x16 v) {   // exact bf16 negation: flip sign bits
  FragU f; f.v = v;
  f.q[0].x ^= 0x80008000u; f.q[0].y ^= 0x80008000u; f.q[0].z ^= 0x80008000u; f.q[0].w ^= 0x80008000u;
  f.q[1].x ^= 0x80008000u; f.q[1].y ^= 0x80008000u; f.q[1].z ^= 0x80008000u; f.q[1].w ^= 0x80008000u;
  return f.v;
}

__device__ __forceinline__ f32x8 wmma_bf16(bf16x16 a, bf16x16 b, f32x8 c) {
  return __builtin_amdgcn_wmma_f32_16x16x32_bf16(false, a, false, b, (short)0, c, false, false);
}

// ---------------- Kernel 1a: w_full (normalize, mirror, pad, bf16) ----------------
__global__ __launch_bounds__(256) void k_wfull(const float* __restrict__ w_re,
                                               const float* __restrict__ w_im,
                                               unsigned short* __restrict__ wfull) {
  int id = blockIdx.x * 256 + threadIdx.x;          // 128*1024 threads
  int r = id >> 10, a = id & 1023;
  if (r < 61) {
    float re = w_re[r * 1024 + a], im = w_im[r * 1024 + a];
    float m = sqrtf(re * re + im * im);
    float s = 1.0f / fmaxf(m, 1.0f);
    unsigned hr = f2bf(re * s), hi = f2bf(im * s);
    wfull[r * 1024 + a]          = (unsigned short)hr;
    wfull[131072 + r * 1024 + a] = (unsigned short)hi;
    if (r < 60) {                                   // w_full[120-r][1023-a] = wn[r][a]
      int r2 = 120 - r, a2 = 1023 - a;
      wfull[r2 * 1024 + a2]          = (unsigned short)hr;
      wfull[131072 + r2 * 1024 + a2] = (unsigned short)hi;
    }
  } else if (r >= 121) {                            // zero pad rows 121..127
    wfull[r * 1024 + a] = 0;
    wfull[131072 + r * 1024 + a] = 0;
  }
}

// ---------------- Kernel 1b: svpT[ri][n][k] = (theta_vec[k] * sv[k][n]) bf16 ----------------
__global__ __launch_bounds__(256) void k_svpt(const float* __restrict__ theta_re,
                                              const float* __restrict__ theta_im,
                                              const float* __restrict__ sv_re,
                                              const float* __restrict__ sv_im,
                                              unsigned short* __restrict__ svpT) {
  int e = blockIdx.x * 256 + threadIdx.x;           // 128 * 2048 threads
  int n = e >> 11;
  int l0 = (e & 2047) * 8;
  unsigned hre[8], him[8];
#pragma unroll
  for (int t = 0; t < 8; ++t) {
    int l = l0 + t;
    int i = l >> 7, j = l & 127;
    int si = (i < 64) ? i : 127 - i;
    int sj = (j < 64) ? j : 127 - j;
    int idx = si * 64 + sj;
    float tr = theta_re[idx], ti = theta_im[idx];
    float inv = rsqrtf(tr * tr + ti * ti);
    float tnr = tr * inv, tni = ti * inv;
    float pr = 0.f, pi = 0.f;
    if (n < 121) {
      float svr = sv_re[(size_t)l * 121 + n];
      float svi = sv_im[(size_t)l * 121 + n];
      pr = tnr * svr - tni * svi;
      pi = tnr * svi + tni * svr;
    }
    hre[t] = f2bf(pr);
    him[t] = f2bf(pi);
  }
  uint4 qr, qi;
  qr.x = hre[0] | (hre[1] << 16); qr.y = hre[2] | (hre[3] << 16);
  qr.z = hre[4] | (hre[5] << 16); qr.w = hre[6] | (hre[7] << 16);
  qi.x = him[0] | (him[1] << 16); qi.y = him[2] | (him[3] << 16);
  qi.z = him[4] | (him[5] << 16); qi.w = him[6] | (him[7] << 16);
  *(uint4*)(svpT + (size_t)n * 16384 + l0)         = qr;
  *(uint4*)(svpT + (size_t)(128 + n) * 16384 + l0) = qi;
}

// ---------------- tile movers for GEMM1 ----------------
__device__ __forceinline__ void g1_loadA(const float* __restrict__ G_re, const float* __restrict__ G_im,
                                         int mtile, int kbase, int tid, float4* ga) {
#pragma unroll
  for (int i = 0; i < 4; ++i) {
    int idx = tid + i * 256;
    int mat = idx >> 9, rem = idx & 511;
    int row = rem >> 3, j = rem & 7;
    const float* Gp = mat ? G_im : G_re;
    ga[i] = *(const float4*)(Gp + (size_t)(mtile * 64 + row) * 16384 + kbase + j * 4);
  }
}
__device__ __forceinline__ void g1_storeA(unsigned short* buf, int tid, const float4* ga) {
#pragma unroll
  for (int i = 0; i < 4; ++i) {
    int idx = tid + i * 256;
    int mat = idx >> 9, rem = idx & 511;
    int row = rem >> 3, j = rem & 7;
    uint2 o;
    o.x = f2bf(ga[i].x) | (f2bf(ga[i].y) << 16);
    o.y = f2bf(ga[i].z) | (f2bf(ga[i].w) << 16);
    *(uint2*)(buf + (mat * A_ROWS + row) * LDS_STRIDE + j * 4) = o;
  }
}
__device__ __forceinline__ void g1_loadB(const unsigned short* __restrict__ svpT, int kbase, int tid, uint4* gb) {
#pragma unroll
  for (int i = 0; i < 4; ++i) {
    int idx = tid + i * 256;
    int mat = idx >> 9, rem = idx & 511;
    int n = rem >> 2, j = rem & 3;
    gb[i] = *(const uint4*)(svpT + (size_t)(mat * 128 + n) * 16384 + kbase + j * 8);
  }
}
__device__ __forceinline__ void gx_storeB(unsigned short* buf, int tid, const uint4* gb) {
#pragma unroll
  for (int i = 0; i < 4; ++i) {
    int idx = tid + i * 256;
    int mat = idx >> 9, rem = idx & 511;
    int n = rem >> 2, j = rem & 3;
    *(uint4*)(buf + LDS_A_HALVES + (mat * B_ROWS + n) * LDS_STRIDE + j * 8) = gb[i];
  }
}

// ---------------- Kernel 2: partial B = G_chunk @ svp_chunk (bf16 WMMA, K-split) ----------------
__global__ __launch_bounds__(256) void k_gemm1(const float* __restrict__ G_re,
                                               const float* __restrict__ G_im,
                                               const unsigned short* __restrict__ svpT,
                                               float* __restrict__ partial) {
  extern __shared__ __align__(16) unsigned short lds[];   // 2 x LDS_TOTAL halves
  int tid   = threadIdx.x;
  int mtile = blockIdx.x >> 3;        // 0..15  (64 rows of G each)
  int kc    = blockIdx.x & 7;         // 0..7   (2048-wide K chunks)
  int lane  = tid & 31;
  int wave  = tid >> 5;
  int strip = wave >> 1;              // 4 M strips of 16 rows
  int nh    = wave & 1;               // 2 N halves of 64 cols

  f32x8 c_re[4] = {}, c_im[4] = {};
  float4 ga[4]; uint4 gb[4];

  // prologue: stage step 0 into buffer 0
  g1_loadA(G_re, G_im, mtile, kc * 2048, tid, ga);
  g1_loadB(svpT, kc * 2048, tid, gb);
  g1_storeA(lds, tid, ga);
  gx_storeB(lds, tid, gb);
  __syncthreads();

  for (int step = 0; step < 64; ++step) {
    unsigned short* cur = lds + (step & 1) * LDS_TOTAL;
    unsigned short* nxt = lds + ((step + 1) & 1) * LDS_TOTAL;
    // issue next tile's global loads early; they complete under the WMMAs below
    if (step + 1 < 64) {
      int kb = kc * 2048 + (step + 1) * 32;
      g1_loadA(G_re, G_im, mtile, kb, tid, ga);
      g1_loadB(svpT, kb, tid, gb);
    }
    // prefetch the G stream two steps ahead (global_prefetch_b8)
    if (step + 2 < 64) {
      int kb2 = kc * 2048 + (step + 2) * 32;
      {
        int row = (tid & 511) >> 3, j = tid & 7;
        __builtin_prefetch(G_re + (size_t)(mtile * 64 + row) * 16384 + kb2 + j * 4, 0, 0);
      }
      {
        int rem = (tid + 512) & 511;
        int row = rem >> 3, j = rem & 7;
        __builtin_prefetch(G_im + (size_t)(mtile * 64 + row) * 16384 + kb2 + j * 4, 0, 0);
      }
    }
    // compute on current buffer
    bf16x16 a_re  = ld_fragA(cur, 0, strip, lane);
    bf16x16 a_im  = ld_fragA(cur, 1, strip, lane);
    bf16x16 a_imn = neg_frag(a_im);
#pragma unroll
    for (int nt = 0; nt < 4; ++nt) {
      bf16x16 b_re = ld_fragB(cur, 0, nh * 64 + nt * 16, lane);
      bf16x16 b_im = ld_fragB(cur, 1, nh * 64 + nt * 16, lane);
      c_re[nt] = wmma_bf16(a_re,  b_re, c_re[nt]);
      c_re[nt] = wmma_bf16(a_imn, b_im, c_re[nt]);   // re -= im*im'
      c_im[nt] = wmma_bf16(a_re,  b_im, c_im[nt]);
      c_im[nt] = wmma_bf16(a_im,  b_re, c_im[nt]);
    }
    // fill the other buffer for the next step
    if (step + 1 < 64) {
      g1_storeA(nxt, tid, ga);
      gx_storeB(nxt, tid, gb);
    }
    __syncthreads();
  }

  // epilogue: partial[ri][kc][n][m] (layout chosen so the reduce is coalesced)
#pragma unroll
  for (int nt = 0; nt < 4; ++nt) {
    int n = nh * 64 + nt * 16 + (lane & 15);
#pragma unroll
    for (int r = 0; r < 8; ++r) {
      int mg = mtile * 64 + strip * 16 + r + ((lane >> 4) << 3);
      partial[((size_t)(0 * 8 + kc) * 128 + n) * 1024 + mg] = c_re[nt][r];
      partial[((size_t)(1 * 8 + kc) * 128 + n) * 1024 + mg] = c_im[nt][r];
    }
  }
}

// ---------------- Kernel 3a: deterministic K-split reduce -> Bt bf16 [2][128][1024] ----------------
__global__ __launch_bounds__(256) void k_reduce(const float* __restrict__ partial,
                                                unsigned short* __restrict__ Bt) {
  int e = blockIdx.x * 256 + threadIdx.x;   // 2*128*1024 threads
  int ri = e >> 17;
  int rem = e & 131071;
  int n = rem >> 10, k = rem & 1023;
  float s = 0.f;
#pragma unroll
  for (int c = 0; c < 8; ++c)
    s += partial[((size_t)(ri * 8 + c) * 128 + n) * 1024 + k];
  Bt[((size_t)ri * 128 + n) * 1024 + k] = (unsigned short)f2bf(s);
}

// ---------------- tile movers for GEMM2 ----------------
__device__ __forceinline__ void g2_loadA(const unsigned short* __restrict__ wfull,
                                         int mtile, int kbase, int tid, uint4* ga2) {
#pragma unroll
  for (int i = 0; i < 2; ++i) {
    int idx = tid + i * 256;
    int mat = idx >> 8, rem = idx & 255;
    int row = rem >> 2, j = rem & 3;
    ga2[i] = *(const uint4*)(wfull + (size_t)(mat * 128 + mtile * 64 + row) * 1024 + kbase + j * 8);
  }
}
__device__ __forceinline__ void g2_storeA(unsigned short* buf, int tid, const uint4* ga2) {
#pragma unroll
  for (int i = 0; i < 2; ++i) {
    int idx = tid + i * 256;
    int mat = idx >> 8, rem = idx & 255;
    int row = rem >> 2, j = rem & 3;
    *(uint4*)(buf + (mat * A_ROWS + row) * LDS_STRIDE + j * 8) = ga2[i];
  }
}
__device__ __forceinline__ void g2_loadB(const unsigned short* __restrict__ Bt, int kbase, int tid, uint4* gb) {
#pragma unroll
  for (int i = 0; i < 4; ++i) {
    int idx = tid + i * 256;
    int mat = idx >> 9, rem = idx & 511;
    int n = rem >> 2, j = rem & 3;
    gb[i] = *(const uint4*)(Bt + (size_t)(mat * 128 + n) * 1024 + kbase + j * 8);
  }
}

// ---------------- Kernel 3b: sout = |w_full @ B|^2 / (A*L)  (bf16 WMMA) ----------------
__global__ __launch_bounds__(256) void k_gemm2(const unsigned short* __restrict__ wfull,
                                               const unsigned short* __restrict__ Bt,
                                               float* __restrict__ sout) {
  extern __shared__ __align__(16) unsigned short lds[];   // 2 x LDS_TOTAL halves
  int tid   = threadIdx.x;
  int mtile = blockIdx.x;             // 0..1
  int lane  = tid & 31;
  int wave  = tid >> 5;
  int strip = wave >> 1;
  int nh    = wave & 1;

  f32x8 c_re[4] = {}, c_im[4] = {};
  uint4 ga2[2]; uint4 gb[4];

  g2_loadA(wfull, mtile, 0, tid, ga2);
  g2_loadB(Bt, 0, tid, gb);
  g2_storeA(lds, tid, ga2);
  gx_storeB(lds, tid, gb);
  __syncthreads();

  for (int step = 0; step < 32; ++step) {
    unsigned short* cur = lds + (step & 1) * LDS_TOTAL;
    unsigned short* nxt = lds + ((step + 1) & 1) * LDS_TOTAL;
    if (step + 1 < 32) {
      int kb = (step + 1) * 32;
      g2_loadA(wfull, mtile, kb, tid, ga2);
      g2_loadB(Bt, kb, tid, gb);
    }
    bf16x16 a_re  = ld_fragA(cur, 0, strip, lane);
    bf16x16 a_im  = ld_fragA(cur, 1, strip, lane);
    bf16x16 a_imn = neg_frag(a_im);
#pragma unroll
    for (int nt = 0; nt < 4; ++nt) {
      bf16x16 b_re = ld_fragB(cur, 0, nh * 64 + nt * 16, lane);
      bf16x16 b_im = ld_fragB(cur, 1, nh * 64 + nt * 16, lane);
      c_re[nt] = wmma_bf16(a_re,  b_re, c_re[nt]);
      c_re[nt] = wmma_bf16(a_imn, b_im, c_re[nt]);
      c_im[nt] = wmma_bf16(a_re,  b_im, c_im[nt]);
      c_im[nt] = wmma_bf16(a_im,  b_re, c_im[nt]);
    }
    if (step + 1 < 32) {
      g2_storeA(nxt, tid, ga2);
      gx_storeB(nxt, tid, gb);
    }
    __syncthreads();
  }

  const float invAL = 1.0f / (1024.0f * 16384.0f);
#pragma unroll
  for (int nt = 0; nt < 4; ++nt) {
    int n = nh * 64 + nt * 16 + (lane & 15);
#pragma unroll
    for (int r = 0; r < 8; ++r) {
      int mg = mtile * 64 + strip * 16 + r + ((lane >> 4) << 3);
      float re = c_re[nt][r];
      float im = c_im[nt][r];
      sout[mg * 128 + n] = (re * re + im * im) * invAL;
    }
  }
}

// ---------------- Kernel 4: diag / band / unbiased std -> scalar loss ----------------
__global__ __launch_bounds__(128) void k_stats(const float* __restrict__ sout,
                                               float* __restrict__ out) {
  __shared__ float s_tot[128], s_band[128], s_d[128], s_y[128];
  int i = threadIdx.x;
  if (i < 121) {
    float tot = 0.f, band = 0.f;
    for (int j = 0; j < 121; ++j) {
      float v = sout[i * 128 + j];
      tot += v;
      int dd = i - j; if (dd < 0) dd = -dd;
      if (dd <= 5) band += v;
    }
    float d = sout[i * 128 + i];
    float x = (float)(i - 60);
    float wgt = 1.0f - 0.2f * x * x / 3600.0f;
    s_tot[i] = tot; s_band[i] = band; s_d[i] = d; s_y[i] = d / wgt;
  }
  __syncthreads();
  if (i == 0) {
    float tt = 0.f, bb = 0.f, dd = 0.f, ym = 0.f;
    for (int k = 0; k < 121; ++k) { tt += s_tot[k]; bb += s_band[k]; dd += s_d[k]; ym += s_y[k]; }
    float mean = ym / 121.0f;
    float var = 0.f;
    for (int k = 0; k < 121; ++k) { float t = s_y[k] - mean; var += t * t; }
    var /= 120.0f;                           // ddof=1
    float amp_loss = -dd + 0.1f * (tt - bb);
    out[0] = amp_loss + 200.0f * sqrtf(var);
  }
}

// ---------------- Host-side launch ----------------
extern "C" void kernel_launch(void* const* d_in, const int* in_sizes, int n_in,
                              void* d_out, int out_size, void* d_ws, size_t ws_size,
                              hipStream_t stream) {
  const float* w_re     = (const float*)d_in[0];
  const float* w_im     = (const float*)d_in[1];
  const float* theta_re = (const float*)d_in[2];
  const float* theta_im = (const float*)d_in[3];
  const float* G_re     = (const float*)d_in[4];
  const float* G_im     = (const float*)d_in[5];
  const float* sv_re    = (const float*)d_in[6];
  const float* sv_im    = (const float*)d_in[7];

  char* ws = (char*)d_ws;
  unsigned short* svpT  = (unsigned short*)(ws + OFF_SVPT);
  unsigned short* wfull = (unsigned short*)(ws + OFF_WFULL);
  float*          part  = (float*)(ws + OFF_PART);
  unsigned short* Bt    = (unsigned short*)(ws + OFF_BT);
  float*          sout  = (float*)(ws + OFF_SOUT);

  k_wfull <<<512, 256, 0, stream>>>(w_re, w_im, wfull);
  k_svpt  <<<1024, 256, 0, stream>>>(theta_re, theta_im, sv_re, sv_im, svpT);
  k_gemm1 <<<128, 256, LDS_BYTES_2BUF, stream>>>(G_re, G_im, svpT, part);
  k_reduce<<<1024, 256, 0, stream>>>(part, Bt);
  k_gemm2 <<<2, 256, LDS_BYTES_2BUF, stream>>>(wfull, Bt, sout);
  k_stats <<<1, 128, 0, stream>>>(sout, (float*)d_out);
}
